// GraphTransformerLayer_62251255988760
// MI455X (gfx1250) — matrix operand order repaired and checked by reading, not compiled, not run on hardware
//
#include <hip/hip_runtime.h>
#include <hip/hip_bf16.h>

// Problem constants (from the reference): B=32, T=128, N=42, E=256, H=8, DH=32
#define BT_TOTAL 4096
#define Nn 42
#define NP 48          // node dim padded to 3x16 WMMA tiles
#define Ee 256
#define Hh 8
#define DHd 32
#define XS 264         // padded LDS row stride (halves) for the 48x256 activation tile
#define SJ 64          // padded j-extent of score rows (2x K=32 WMMA steps)
#define WP_HALVES (4 * 16 * 8 * 32 * 16)   // 4 mats x 16 ntiles x 8 ksteps x 32 lanes x 16 halves
#define WP_BYTES  (WP_HALVES * 2)          // 512 KB

typedef __attribute__((ext_vector_type(16))) _Float16 v16h;
typedef __attribute__((ext_vector_type(8)))  _Float16 v8h;
typedef __attribute__((ext_vector_type(8)))  float    v8f;

union V16 { v16h v; v8h h[2]; _Float16 e[16]; };

__device__ __forceinline__ v8f wmma_f16(v16h a, v16h b, v8f c) {
  // D = A(16x32 f16) * B(32x16 f16) + C(16x16 f32)
  return __builtin_amdgcn_wmma_f32_16x16x32_f16(false, a, false, b, (short)0, c,
                                                false, false);
}

// A fragment (16x32 f16) from a row-major f16 LDS matrix.
// Lanes 0-15 / 16-31 both hold rows m0..m0+15; lane-half selects the K sub-block:
// 16 contiguous K halves, K(base..base+15), base = 16*lanehalf.
__device__ __forceinline__ v16h lda(const _Float16* base, int stride, int m0, int k0,
                                    int lane) {
  int row = m0 + (lane & 15);
  int lh  = (lane >> 4) & 1;
  const _Float16* p = base + row * stride + k0 + lh * 16;
  V16 r;
  r.h[0] = *(const v8h*)(p);        // 16B LDS read
  r.h[1] = *(const v8h*)(p + 8);    // 16B LDS read
  return r.v;
}

// B fragment (32x16 f16) where the matrix is stored as Bt[n][k] row-major
// (i.e. B[k][n] column-major). Lane = column n, 16 contiguous K halves,
// lane-half selects K base 0/16.
__device__ __forceinline__ v16h ldb(const _Float16* base, int stride, int n0, int k0,
                                    int lane) {
  int col = n0 + (lane & 15);
  int lh  = (lane >> 4) & 1;
  const _Float16* p = base + col * stride + k0 + lh * 16;
  V16 r;
  r.h[0] = *(const v8h*)(p);
  r.h[1] = *(const v8h*)(p + 8);
  return r.v;
}

// Fallback: B fragment (32x16) gathered from a row-major fp32 weight W[k][n]
// in global (L2-resident), converted to f16 on the fly.
__device__ __forceinline__ v16h ldw(const float* __restrict__ W, int k0, int n0,
                                    int lane) {
  int col = n0 + (lane & 15);
  int kb  = k0 + ((lane & 16) ? 16 : 0);
  const float* p = W + (size_t)kb * Ee + col;
  V16 r;
#pragma unroll
  for (int e2 = 0; e2 < 16; ++e2) r.e[e2] = (_Float16)p[(size_t)e2 * Ee];
  return r.v;
}

// Prepacked: fragment `frag` = (mat*16 + ntile)*8 + kstep; each lane reads its
// 16 halves contiguously (32B/lane, 1KB per wave -> fully coalesced b128 pair).
__device__ __forceinline__ v16h ldwp(const _Float16* __restrict__ Wp, int frag,
                                     int lane) {
  const _Float16* p = Wp + (((size_t)frag * 32 + lane) << 4);
  V16 r;
  r.h[0] = *(const v8h*)(p);
  r.h[1] = *(const v8h*)(p + 8);
  return r.v;
}

// One-shot weight prepack: Wp[mat][nt][ks][lane][e] = f16(W[k][n]) with
// k = ks*32 + 16*(lane>=16) + e, n = nt*16 + (lane&15).
__global__ __launch_bounds__(256) void prepack_kernel(
    const float* __restrict__ Wq, const float* __restrict__ Wk,
    const float* __restrict__ Wv, const float* __restrict__ Wo,
    _Float16* __restrict__ Wp) {
  int idx = blockIdx.x * 256 + threadIdx.x;    // 0 .. WP_HALVES-1
  if (idx >= WP_HALVES) return;
  int e    = idx & 15;
  int lane = (idx >> 4) & 31;
  int ks   = (idx >> 9) & 7;
  int nt   = (idx >> 12) & 15;
  int mat  = idx >> 16;
  const float* W = (mat == 0) ? Wq : (mat == 1) ? Wk : (mat == 2) ? Wv : Wo;
  int k = ks * 32 + ((lane & 16) ? 16 : 0) + e;
  int n = nt * 16 + (lane & 15);
  Wp[idx] = (_Float16)W[k * Ee + n];
}

template <bool PRE>
__global__ __launch_bounds__(256) void graph_attn_kernel(
    const float* __restrict__ X,      // (B,T,N,E)
    const int*   __restrict__ dist,   // (N,N) int32
    const float* __restrict__ Wq, const float* __restrict__ bq,
    const float* __restrict__ Wk, const float* __restrict__ bk,
    const float* __restrict__ Wv, const float* __restrict__ bv,
    const float* __restrict__ Wo, const float* __restrict__ bo,
    const float* __restrict__ rel,    // (15,H)
    const _Float16* __restrict__ Wp,  // prepacked weights (or nullptr)
    float* __restrict__ out,          // (B,T,N,E)
    float* __restrict__ attn)         // (B,T,H,N,N)
{
  // ~156 KB static LDS (one workgroup per WGP-half; 320 KB per WGP on CDNA5)
  __shared__ __attribute__((aligned(16))) _Float16 sX[NP * XS];        // X, later O
  __shared__ __attribute__((aligned(16))) _Float16 sQ[Hh * NP * DHd];  // [h][node][d]
  __shared__ __attribute__((aligned(16))) _Float16 sK[Hh * NP * DHd];  // [h][node][d]
  __shared__ __attribute__((aligned(16))) _Float16 sVt[Hh * DHd * SJ]; // [h][d][node]
  __shared__ __attribute__((aligned(16))) _Float16 sS[Hh * NP * SJ];   // [h][i][j]
  __shared__ unsigned char sDist[Nn * Nn];
  __shared__ float sRel[15 * Hh];

  const int tid  = threadIdx.x;
  const int lane = tid & 31;
  const int wave = tid >> 5;
  const int bt   = blockIdx.x;      // (b*T + t), 0..4095

  // ---- Stage 0: stage X tile to LDS as f16 (pad rows zeroed); zero pad buffers ----
  const float* Xg = X + (size_t)bt * (Nn * Ee);
  for (int idx = tid; idx < NP * Ee; idx += 256) {
    int r = idx >> 8, c = idx & 255;
    sX[r * XS + c] = (r < Nn) ? (_Float16)Xg[r * Ee + c] : (_Float16)0.0f;
  }
  for (int idx = tid; idx < Hh * DHd * SJ; idx += 256) sVt[idx] = (_Float16)0.0f;
  for (int idx = tid; idx < Hh * NP * SJ;  idx += 256) sS[idx]  = (_Float16)0.0f;
  for (int idx = tid; idx < Nn * Nn; idx += 256) sDist[idx] = (unsigned char)dist[idx];
  for (int idx = tid; idx < 15 * Hh; idx += 256) sRel[idx] = rel[idx];
  __syncthreads();

  // ---- Stage 1: Q,K,V projections. 48 strips = {Q,K,V} x 16 n-tiles.
  // One weight B-fragment per K-step, reused across 3 M-tiles. ----
  for (int s = wave; s < 48; s += 8) {
    int mat = s >> 4, nt = s & 15;
    const float* W    = (mat == 0) ? Wq : (mat == 1) ? Wk : Wv;
    const float* bvec = (mat == 0) ? bq : (mat == 1) ? bk : bv;
    v8f a0 = {0.f,0.f,0.f,0.f,0.f,0.f,0.f,0.f};
    v8f a1 = a0, a2 = a0;
    for (int k0 = 0; k0 < Ee; k0 += 32) {
      v16h b;
      if (PRE) {
        b = ldwp(Wp, ((mat * 16 + nt) << 3) + (k0 >> 5), lane);
      } else {
        if (k0 + 32 < Ee) __builtin_prefetch(W + (size_t)(k0 + 32) * Ee + nt * 16, 0, 0);
        b = ldw(W, k0, nt * 16, lane);
      }
      a0 = wmma_f16(lda(sX, XS,  0, k0, lane), b, a0);
      a1 = wmma_f16(lda(sX, XS, 16, k0, lane), b, a1);
      a2 = wmma_f16(lda(sX, XS, 32, k0, lane), b, a2);
    }
    int c = nt * 16 + (lane & 15);
    float bias = bvec[c];
    int h = c >> 5, d = c & 31;
    int rb = (lane & 16) ? 8 : 0;
#pragma unroll
    for (int r = 0; r < 8; ++r) {
      float v0 = a0[r] + bias, v1 = a1[r] + bias, v2 = a2[r] + bias;
      int r0 = r + rb, r1 = 16 + r + rb, r2 = 32 + r + rb;
      if (mat == 0) {
        sQ[(h * NP + r0) * DHd + d] = (_Float16)v0;
        sQ[(h * NP + r1) * DHd + d] = (_Float16)v1;
        sQ[(h * NP + r2) * DHd + d] = (_Float16)v2;
      } else if (mat == 1) {
        sK[(h * NP + r0) * DHd + d] = (_Float16)v0;
        sK[(h * NP + r1) * DHd + d] = (_Float16)v1;
        sK[(h * NP + r2) * DHd + d] = (_Float16)v2;
      } else {  // V stored transposed [h][d][node] for contiguous B-fragment loads
        sVt[(h * DHd + d) * SJ + r0] = (_Float16)v0;
        sVt[(h * DHd + d) * SJ + r1] = (_Float16)v1;
        sVt[(h * DHd + d) * SJ + r2] = (_Float16)v2;
      }
    }
  }
  __syncthreads();

  // ---- Stage 2: scores = Q K^T / sqrt(DH) + rel_bias, adjacency-masked ----
  const float inv_sqrt = 0.17677669529663687f;   // 1/sqrt(32)
  for (int u = wave; u < 24; u += 8) {
    int h = u / 3, mt = u % 3;
    const _Float16* Qh = sQ + h * NP * DHd;
    const _Float16* Kh = sK + h * NP * DHd;  // row-major [node][d] == Kt for B layout
    v16h aq = lda(Qh, DHd, mt * 16, 0, lane);
#pragma unroll
    for (int jt = 0; jt < 3; ++jt) {
      v16h bk16 = ldb(Kh, DHd, jt * 16, 0, lane);
      v8f sc = {0.f,0.f,0.f,0.f,0.f,0.f,0.f,0.f};
      sc = wmma_f16(aq, bk16, sc);
      int j  = jt * 16 + (lane & 15);
      int rb = (lane & 16) ? 8 : 0;
#pragma unroll
      for (int r = 0; r < 8; ++r) {
        int i = mt * 16 + r + rb;
        float v = -30000.0f;                     // masked (exp underflows to 0)
        if (i < Nn && j < Nn) {
          int dd = sDist[i * Nn + j];            // edge <=> shortest distance == 1
          if (dd == 1) v = sc[r] * inv_sqrt + sRel[dd * Hh + h];
        }
        sS[(h * NP + i) * SJ + j] = (_Float16)v;
      }
    }
  }
  __syncthreads();

  // ---- Stage 3: softmax over j (wave per row, shfl reductions);
  //      write attn (coalesced f32) + normalized f16 back to LDS ----
  for (int rw = wave; rw < Hh * Nn; rw += 8) {
    int h = rw / Nn, i = rw - h * Nn;
    _Float16* row = sS + (h * NP + i) * SJ;
    float v0 = (float)row[lane];                               // j = lane
    float v1 = (lane < 16) ? (float)row[32 + lane] : -30000.0f; // j = 32+lane (<48)
    float mx = fmaxf(v0, v1);
#pragma unroll
    for (int off = 16; off > 0; off >>= 1) mx = fmaxf(mx, __shfl_xor(mx, off));
    float e0 = __expf(v0 - mx);
    float e1 = (lane < 16) ? __expf(v1 - mx) : 0.0f;
    float sm = e0 + e1;
#pragma unroll
    for (int off = 16; off > 0; off >>= 1) sm += __shfl_xor(sm, off);
    float rs = 1.0f / sm;
    float a0 = e0 * rs, a1 = e1 * rs;
    row[lane]      = (_Float16)a0;     // cols 0..31
    row[32 + lane] = (_Float16)a1;     // cols 32..63 (pad cols get exact 0)
    float* ag = attn + (((size_t)bt * Hh + h) * Nn + i) * Nn;
    ag[lane] = a0;                               // j = 0..31  (coalesced)
    if (lane < Nn - 32) ag[32 + lane] = a1;      // j = 32..41 (coalesced)
  }
  __syncthreads();

  // ---- Stage 4: O = attn @ V  (per head: 3 m-tiles x 2 d-tiles, K=64 padded j) ----
  for (int u = wave; u < 48; u += 8) {
    int h = u / 6, rest = u % 6, mt = rest >> 1, dt = rest & 1;
    const _Float16* Sh = sS  + h * NP * SJ;
    const _Float16* Vh = sVt + h * DHd * SJ;
    v8f acc = {0.f,0.f,0.f,0.f,0.f,0.f,0.f,0.f};
#pragma unroll
    for (int kk = 0; kk < 2; ++kk) {
      acc = wmma_f16(lda(Sh, SJ, mt * 16, kk * 32, lane),
                     ldb(Vh, SJ, dt * 16, kk * 32, lane), acc);
    }
    int c  = h * DHd + dt * 16 + (lane & 15);
    int rb = (lane & 16) ? 8 : 0;
#pragma unroll
    for (int r = 0; r < 8; ++r)
      sX[(mt * 16 + r + rb) * XS + c] = (_Float16)acc[r];   // sX reused as O
  }
  __syncthreads();

  // ---- Stage 5: out = O @ Wo + bo ----
  for (int s2 = wave; s2 < 16; s2 += 8) {
    int nt = s2;
    v8f a0 = {0.f,0.f,0.f,0.f,0.f,0.f,0.f,0.f};
    v8f a1 = a0, a2 = a0;
    for (int k0 = 0; k0 < Ee; k0 += 32) {
      v16h b;
      if (PRE) {
        b = ldwp(Wp, ((3 * 16 + nt) << 3) + (k0 >> 5), lane);
      } else {
        if (k0 + 32 < Ee) __builtin_prefetch(Wo + (size_t)(k0 + 32) * Ee + nt * 16, 0, 0);
        b = ldw(Wo, k0, nt * 16, lane);
      }
      a0 = wmma_f16(lda(sX, XS,  0, k0, lane), b, a0);
      a1 = wmma_f16(lda(sX, XS, 16, k0, lane), b, a1);
      a2 = wmma_f16(lda(sX, XS, 32, k0, lane), b, a2);
    }
    int c = nt * 16 + (lane & 15);
    float bias = bo[c];
    int rb = (lane & 16) ? 8 : 0;
    float* Og = out + (size_t)bt * (Nn * Ee);
#pragma unroll
    for (int r = 0; r < 8; ++r) {
      int r0 = r + rb;       Og[r0 * Ee + c] = a0[r] + bias;
      int r1 = 16 + r + rb;  Og[r1 * Ee + c] = a1[r] + bias;
      int r2 = 32 + r + rb;  if (r2 < Nn) Og[r2 * Ee + c] = a2[r] + bias;
    }
  }
}

extern "C" void kernel_launch(void* const* d_in, const int* in_sizes, int n_in,
                              void* d_out, int out_size, void* d_ws, size_t ws_size,
                              hipStream_t stream) {
  (void)in_sizes; (void)n_in; (void)out_size;
  const float* X    = (const float*)d_in[0];
  // d_in[1] = adjacency (unused: mask == (dist==1))
  const int*   dist = (const int*)d_in[2];
  const float* Wq   = (const float*)d_in[3];
  const float* bq   = (const float*)d_in[4];
  const float* Wk   = (const float*)d_in[5];
  const float* bk   = (const float*)d_in[6];
  const float* Wv   = (const float*)d_in[7];
  const float* bv   = (const float*)d_in[8];
  const float* Wo   = (const float*)d_in[9];
  const float* bo   = (const float*)d_in[10];
  const float* rel  = (const float*)d_in[11];
  float* out  = (float*)d_out;
  float* attn = out + (size_t)BT_TOTAL * Nn * Ee;  // out then attn, concatenated

  if (ws_size >= (size_t)WP_BYTES) {
    _Float16* Wp = (_Float16*)d_ws;
    prepack_kernel<<<dim3((WP_HALVES + 255) / 256), dim3(256), 0, stream>>>(
        Wq, Wk, Wv, Wo, Wp);
    graph_attn_kernel<true><<<dim3(BT_TOTAL), dim3(256), 0, stream>>>(
        X, dist, Wq, bq, Wk, bk, Wv, bv, Wo, bo, rel, Wp, out, attn);
  } else {
    graph_attn_kernel<false><<<dim3(BT_TOTAL), dim3(256), 0, stream>>>(
        X, dist, Wq, bq, Wk, bk, Wv, bv, Wo, bo, rel, (const _Float16*)nullptr,
        out, attn);
  }
}
